// ImageGraphHyperbolicGATClassifier_52029233824310
// MI455X (gfx1250) — matrix operand order repaired and checked by reading, not compiled, not run on hardware
//
#include <hip/hip_runtime.h>
#include <cstddef>

// ---------------------------------------------------------------------------
// CDNA5 (gfx1250) vector types
// ---------------------------------------------------------------------------
typedef __bf16        v16bf __attribute__((ext_vector_type(16)));
typedef float         v8f   __attribute__((ext_vector_type(8)));
typedef unsigned int  u32x4 __attribute__((ext_vector_type(4)));
typedef unsigned int  v4u   __attribute__((ext_vector_type(4)));
typedef int           v4i   __attribute__((ext_vector_type(4)));
typedef int           v8i   __attribute__((ext_vector_type(8)));

union BfPack {
  u32x4 u[2];   // two 16B chunks = 16 bf16
  v16bf v;
};

#if __has_builtin(__builtin_amdgcn_tensor_load_to_lds) && \
    __has_builtin(__builtin_amdgcn_s_wait_tensorcnt)
#define HAVE_TDM 1
#else
#define HAVE_TDM 0
#endif

// ---------------------------------------------------------------------------
// Helpers
// ---------------------------------------------------------------------------
__device__ __forceinline__ unsigned short f2bf(float f) {
  unsigned u = __float_as_uint(f);
  unsigned r = u + 0x7FFFu + ((u >> 16) & 1u);   // round-to-nearest-even
  return (unsigned short)(r >> 16);
}

// float -> totally-ordered u32 key (for atomicMax-based segment max)
__device__ __forceinline__ unsigned fkey(float f) {
  unsigned b = __float_as_uint(f);
  return b ^ ((unsigned)((int)b >> 31) | 0x80000000u);
}
__device__ __forceinline__ float funkey(unsigned k) {
  unsigned b = (k & 0x80000000u) ? (k ^ 0x80000000u) : ~k;
  return __uint_as_float(b);
}

#define NEG_INF_KEY 0x007FFFFFu   /* fkey(-inf) */

// ---------------------------------------------------------------------------
// Fill / convert kernels
// ---------------------------------------------------------------------------
__global__ void fill_f32(float* p, float v, size_t n) {
  size_t i = (size_t)blockIdx.x * blockDim.x + threadIdx.x;
  size_t s = (size_t)gridDim.x * blockDim.x;
  for (; i < n; i += s) p[i] = v;
}
__global__ void fill_u32(unsigned* p, unsigned v, size_t n) {
  size_t i = (size_t)blockIdx.x * blockDim.x + threadIdx.x;
  size_t s = (size_t)gridDim.x * blockDim.x;
  for (; i < n; i += s) p[i] = v;
}
__global__ void f32_to_bf16(const float* __restrict__ src, unsigned short* __restrict__ dst, size_t n) {
  size_t i = (size_t)blockIdx.x * blockDim.x + threadIdx.x;
  size_t s = (size_t)gridDim.x * blockDim.x;
  for (; i < n; i += s) dst[i] = f2bf(src[i]);
}

// ---------------------------------------------------------------------------
// LayerNorm (+optional additive bias, +optional ReLU), output f32 or bf16.
// One block per row, blockDim = 256 (8 wave32's).
// ---------------------------------------------------------------------------
__global__ void ln_kernel(const float* __restrict__ X, const float* __restrict__ addb,
                          const float* __restrict__ w, const float* __restrict__ b,
                          int len, int relu,
                          float* __restrict__ of, unsigned short* __restrict__ ob) {
  const float* row = X + (size_t)blockIdx.x * len;
  float s = 0.f, s2 = 0.f;
  for (int c = threadIdx.x; c < len; c += blockDim.x) {
    float v = row[c] + (addb ? addb[c] : 0.f);
    s += v; s2 += v * v;
  }
  for (int o = 16; o > 0; o >>= 1) { s += __shfl_down(s, o); s2 += __shfl_down(s2, o); }
  __shared__ float rs[8], rs2[8];
  __shared__ float mu_sh, rstd_sh;
  int wid = threadIdx.x >> 5, ln = threadIdx.x & 31;
  if (ln == 0) { rs[wid] = s; rs2[wid] = s2; }
  __syncthreads();
  if (threadIdx.x == 0) {
    float ts = 0.f, ts2 = 0.f;
    int nw = blockDim.x >> 5;
    for (int i = 0; i < nw; ++i) { ts += rs[i]; ts2 += rs2[i]; }
    float mu  = ts / (float)len;
    float var = ts2 / (float)len - mu * mu;
    mu_sh = mu;
    rstd_sh = rsqrtf(var + 1e-5f);
  }
  __syncthreads();
  float mu = mu_sh, rstd = rstd_sh;
  for (int c = threadIdx.x; c < len; c += blockDim.x) {
    float v = row[c] + (addb ? addb[c] : 0.f);
    float y = (v - mu) * rstd * w[c] + b[c];
    if (relu) y = fmaxf(y, 0.f);
    if (of) of[(size_t)blockIdx.x * len + c] = y;
    else    ob[(size_t)blockIdx.x * len + c] = f2bf(y);
  }
}

// ---------------------------------------------------------------------------
// bf16 WMMA GEMM:  C[M,ncols] = A[M,KD](bf16) * W[ncols,KD](bf16)^T  (f32 acc)
//
// One block per 16-row strip; ncols == NT * 128; KD compile-time so all tile
// displacements (t * 128*KD elements, max 1.75MB < 8MB) fold into the 24-bit
// instruction offsets.
//
// The 16 x KD A strip (32KB) is staged into LDS once via the Tensor Data
// Mover. The k-loop is manually double-buffered: the load group for k+32 is
// issued in program order BEFORE the WMMA group for k, so each WMMA consumes
// loads issued one full phase earlier (17 loads + NT WMMAs in between) --
// partial waits by construction, and the per-tile accumulator RAW chain only
// recurs every NT independent matrix ops.
//
// A frag: lane(&15)=row, K chunks {klo..klo+7, klo+16..klo+23}, klo=(lane>>4)*8
// B frag: lane(&15)=col, K chunks {klo..klo+7, klo+16..klo+23}, klo=(lane>>4)*8
// C frag: VGPR r -> row r + 8*(lane>>4), col lane&15
// ---------------------------------------------------------------------------
template <int NT, int KD>
__global__ __launch_bounds__(256, 1)
void gemm_bf16_wmma(const unsigned short* __restrict__ A,
                    const unsigned short* __restrict__ W,
                    float* __restrict__ C) {
  constexpr int NCOLS   = NT * 128;
  constexpr size_t TSTR = (size_t)128 * KD;   // element stride between a wave's tiles
  __shared__ __align__(16) unsigned short Asm[16 * KD];
  int lane  = threadIdx.x & 31;
  int wave  = threadIdx.x >> 5;
  int mbase = blockIdx.x << 4;

  // ---- stage the 16 x KD bf16 A strip into LDS ----
#if HAVE_TDM
  if (wave == 0) {
    unsigned long long ga = (unsigned long long)(const void*)(A + (size_t)mbase * KD);
    unsigned ldsoff = (unsigned)(unsigned long long)(void*)&Asm[0];  // LDS byte offset
    // D# group 0 (ISA 8.3): count=1 | lds_addr | global_addr(57b) | type=2
    v4u g0;
    g0[0] = 1u;
    g0[1] = ldsoff;
    g0[2] = (unsigned)(ga & 0xFFFFFFFFu);
    g0[3] = (unsigned)((ga >> 32) & 0x01FFFFFFu) | (2u << 30);
    // D# group 1 (ISA 8.4): data_size=2B; tensor 16 x KD; tile 16 x KD; stride KD
    v8i g1;
    g1[0] = 1 << 16;                                    // wg_mask=0, data_size=1 (2B)
    g1[1] = (KD & 0xFFFF) << 16;                        // tensor_dim0[15:0]
    g1[2] = ((KD >> 16) & 0xFFFF) | (16 << 16);         // tensor_dim0[31:16] | tensor_dim1=16
    g1[3] = (KD & 0xFFFF) << 16;                        // tensor_dim1 hi | tile_dim0=KD
    g1[4] = 16;                                         // tile_dim1=16, tile_dim2=0
    g1[5] = KD;                                         // tensor_dim0_stride[31:0]
    g1[6] = 0;
    g1[7] = 0;
    v4i g2 = {0, 0, 0, 0};
    v4i g3 = {0, 0, 0, 0};
#if __has_include(<hip/amd_detail/amd_gfx1250_TDM.h>)
    v8i g4 = {0, 0, 0, 0, 0, 0, 0, 0};
    __builtin_amdgcn_tensor_load_to_lds(g0, g1, g2, g3, g4, 0);
#else
    __builtin_amdgcn_tensor_load_to_lds(g0, g1, g2, g3, 0);
#endif
    __builtin_amdgcn_s_wait_tensorcnt(0);
  }
  __syncthreads();
#else
  {
    const u32x4* src = reinterpret_cast<const u32x4*>(A + (size_t)mbase * KD);
    u32x4* dst = reinterpret_cast<u32x4*>(Asm);
    int nv = (16 * KD) >> 3;   // u32x4 = 8 bf16
    for (int i = threadIdx.x; i < nv; i += blockDim.x) dst[i] = src[i];
    __syncthreads();
  }
#endif

  const unsigned short* ap = Asm + (size_t)(lane & 15) * KD + ((lane >> 4) << 3);
  // B base for this wave's tile 0; successive tiles are 128 rows (8 waves*16) apart
  const unsigned short* bp =
      W + (size_t)((wave << 4) + (lane & 15)) * KD + ((lane >> 4) << 3);

  v8f acc[NT];
#pragma unroll
  for (int t = 0; t < NT; ++t)
    acc[t] = (v8f){0.f, 0.f, 0.f, 0.f, 0.f, 0.f, 0.f, 0.f};

  auto loadA = [&](BfPack& a, int k) {
    a.u[0] = *reinterpret_cast<const u32x4*>(ap + k);          // ds_load_b128
    a.u[1] = *reinterpret_cast<const u32x4*>(ap + k + 16);
  };
  auto loadB = [&](BfPack* b, int k) {
#pragma unroll
    for (int t = 0; t < NT; ++t) {
      b[t].u[0] = *reinterpret_cast<const u32x4*>(bp + TSTR * t + k);      // global_load_b128
      b[t].u[1] = *reinterpret_cast<const u32x4*>(bp + TSTR * t + k + 16);
    }
  };
  auto mmaAll = [&](const BfPack& a, const BfPack* b) {
#pragma unroll
    for (int t = 0; t < NT; ++t)
      acc[t] = __builtin_amdgcn_wmma_f32_16x16x32_bf16(
          /*neg_a=*/false, a.v, /*neg_b=*/false, b[t].v,
          /*c_mod=*/(short)0, acc[t], /*reuse_a=*/false, /*reuse_b=*/false);
  };

  BfPack a0, a1;
  BfPack b0[NT], b1[NT];
  loadA(a0, 0);
  loadB(b0, 0);
#pragma unroll 1
  for (int k = 0; k < KD - 64; k += 64) {
    loadA(a1, k + 32);
    loadB(b1, k + 32);
    __builtin_prefetch(bp + k + 512, 0, 1);                    // global_prefetch_b8
    mmaAll(a0, b0);          // consumes group k (issued one phase earlier)
    loadA(a0, k + 64);
    loadB(b0, k + 64);
    mmaAll(a1, b1);          // consumes group k+32
  }
  // epilogue: groups KD-64 (already in a0/b0) and KD-32
  loadA(a1, KD - 32);
  loadB(b1, KD - 32);
  mmaAll(a0, b0);
  mmaAll(a1, b1);

  int col0  = (wave << 4) + (lane & 15);
  int rbase = mbase + ((lane >> 4) << 3);
#pragma unroll
  for (int t = 0; t < NT; ++t) {
    int col = col0 + 128 * t;
#pragma unroll
    for (int r = 0; r < 8; ++r)
      C[(size_t)(rbase + r) * NCOLS + col] = acc[t][r];
  }
}

// ---------------------------------------------------------------------------
// Attention coefficients: one wave32 per (node, head)
// ---------------------------------------------------------------------------
__global__ void attn_coef(const float* __restrict__ Hf,
                          const float* __restrict__ asrc, const float* __restrict__ adst,
                          int Nn, int H, int C,
                          float* __restrict__ as_, float* __restrict__ ad_) {
  int w    = (blockIdx.x * blockDim.x + threadIdx.x) >> 5;
  int lane = threadIdx.x & 31;
  if (w >= Nn * H) return;
  int n = w / H, h = w - n * H;
  const float* row = Hf + (size_t)n * (H * C) + h * C;
  float ps = 0.f, pd = 0.f;
  for (int c = lane; c < C; c += 32) {
    float hv = row[c];
    ps += hv * asrc[h * C + c];
    pd += hv * adst[h * C + c];
  }
  for (int o = 16; o > 0; o >>= 1) { ps += __shfl_down(ps, o); pd += __shfl_down(pd, o); }
  if (lane == 0) { as_[w] = ps; ad_[w] = pd; }
}

// ---------------------------------------------------------------------------
// Edge passes (E real edges + Nn self-loops). Segment softmax over dst.
// h / out buffers (2 x 80MB) fit the 192MB L2 -> atomics stay L2-resident.
// ---------------------------------------------------------------------------
__global__ void edge_max(const int* __restrict__ ei, int E, int Nn, int H,
                         const float* __restrict__ as_, const float* __restrict__ ad_,
                         unsigned* __restrict__ mx) {
  int total = (E + Nn) * H;
  for (int idx = blockIdx.x * blockDim.x + threadIdx.x; idx < total;
       idx += gridDim.x * blockDim.x) {
    int e = idx / H, h = idx - e * H;
    int s, d;
    if (e < E) { s = ei[e]; d = ei[E + e]; } else { s = e - E; d = s; }
    float v = as_[s * H + h] + ad_[d * H + h];
    v = v > 0.f ? v : 0.2f * v;                 // leaky_relu
    atomicMax(&mx[d * H + h], fkey(v));
  }
}

__global__ void edge_sum(const int* __restrict__ ei, int E, int Nn, int H,
                         const float* __restrict__ as_, const float* __restrict__ ad_,
                         const unsigned* __restrict__ mx, float* __restrict__ sm) {
  int total = (E + Nn) * H;
  for (int idx = blockIdx.x * blockDim.x + threadIdx.x; idx < total;
       idx += gridDim.x * blockDim.x) {
    int e = idx / H, h = idx - e * H;
    int s, d;
    if (e < E) { s = ei[e]; d = ei[E + e]; } else { s = e - E; d = s; }
    float v = as_[s * H + h] + ad_[d * H + h];
    v = v > 0.f ? v : 0.2f * v;
    atomicAdd(&sm[d * H + h], __expf(v - funkey(mx[d * H + h])));
  }
}

// one wave32 per (edge, head): 128-channel weighted scatter into out[dst]
__global__ void edge_aggr(const int* __restrict__ ei, int E, int Nn, int H, int C,
                          const float* __restrict__ as_, const float* __restrict__ ad_,
                          const unsigned* __restrict__ mx, const float* __restrict__ sm,
                          const float* __restrict__ Hf, float* __restrict__ Of) {
  int w    = (blockIdx.x * blockDim.x + threadIdx.x) >> 5;
  int lane = threadIdx.x & 31;
  int total = (E + Nn) * H;
  if (w >= total) return;
  int e = w / H, h = w - e * H;
  int s, d;
  if (e < E) { s = ei[e]; d = ei[E + e]; } else { s = e - E; d = s; }
  float v = as_[s * H + h] + ad_[d * H + h];
  v = v > 0.f ? v : 0.2f * v;
  float alpha = __expf(v - funkey(mx[d * H + h])) / sm[d * H + h];
  const float* hs = Hf + (size_t)s * (H * C) + h * C;
  float*       od = Of + (size_t)d * (H * C) + h * C;
  for (int c = lane; c < C; c += 32)
    atomicAdd(&od[c], hs[c] * alpha);   // global_atomic_add_f32
}

// ---------------------------------------------------------------------------
// global_mean_pool: one wave32 per node
// ---------------------------------------------------------------------------
__global__ void pool_kernel(const float* __restrict__ hf, const int* __restrict__ batch,
                            float* __restrict__ sums, float* __restrict__ cnt, int Nn) {
  int w    = (blockIdx.x * blockDim.x + threadIdx.x) >> 5;
  int lane = threadIdx.x & 31;
  if (w >= Nn) return;
  int g = batch[w];
  const float* row = hf + (size_t)w * 128;
  float* dr = sums + g * 128;
  for (int c = lane; c < 128; c += 32) atomicAdd(&dr[c], row[c]);
  if (lane == 0) atomicAdd(&cnt[g], 1.0f);
}

// ---------------------------------------------------------------------------
// Poincare head: pooled -> linear(32) -> eval-BN -> relu -> ball clip -> logits
// out[0:112] = logits (16x7), out[112:624] = z (16x32)
// ---------------------------------------------------------------------------
__global__ void head_kernel(const float* __restrict__ sums, const float* __restrict__ cnt,
                            const float* __restrict__ pw, const float* __restrict__ pb,
                            const float* __restrict__ bng, const float* __restrict__ bnb,
                            const float* __restrict__ cw, const float* __restrict__ cb,
                            float* __restrict__ out) {
  __shared__ float P[16 * 128];
  __shared__ float Z[16 * 32];
  __shared__ float nrm[16];
  int t = threadIdx.x;
  for (int i = t; i < 16 * 128; i += blockDim.x) {
    int g = i >> 7;
    P[i] = sums[i] / fmaxf(cnt[g], 1.0f);
  }
  __syncthreads();
  const float bnscale = rsqrtf(1.0f + 1e-5f);
  for (int i = t; i < 16 * 32; i += blockDim.x) {
    int g = i >> 5, j = i & 31;
    float acc = pb[j];
    for (int k = 0; k < 128; ++k) acc += P[g * 128 + k] * pw[j * 128 + k];
    float z = acc * bnscale * bng[j] + bnb[j];
    Z[i] = fmaxf(z, 0.f);
  }
  __syncthreads();
  if (t < 16) {
    float s = 0.f;
    for (int j = 0; j < 32; ++j) { float z = Z[t * 32 + j]; s += z * z; }
    nrm[t] = sqrtf(s);
  }
  __syncthreads();
  for (int i = t; i < 16 * 32; i += blockDim.x) {
    int g = i >> 5;
    float z = Z[i];
    float nv = nrm[g];
    const float mn = 1.0f - 1e-5f;
    if (nv >= mn) z = z / nv * mn;
    Z[i] = z;
    out[112 + i] = z;
  }
  __syncthreads();
  for (int i = t; i < 16 * 7; i += blockDim.x) {
    int g = i / 7, j = i - g * 7;
    float acc = cb[j];
    for (int k = 0; k < 32; ++k) acc += Z[g * 32 + k] * cw[j * 32 + k];
    out[g * 7 + j] = acc;
  }
}

// ---------------------------------------------------------------------------
// Host-side orchestration
// ---------------------------------------------------------------------------
extern "C" void kernel_launch(void* const* d_in, const int* in_sizes, int n_in,
                              void* d_out, int out_size, void* d_ws, size_t ws_size,
                              hipStream_t stream) {
  (void)in_sizes; (void)n_in; (void)out_size; (void)ws_size;

  const float* x       = (const float*)d_in[0];
  const int*   ei      = (const int*)d_in[1];
  const int*   batch   = (const int*)d_in[2];
  const float* ln_in_w = (const float*)d_in[3];
  const float* ln_in_b = (const float*)d_in[4];
  const float* W1      = (const float*)d_in[5];
  const float* a_src1  = (const float*)d_in[6];
  const float* a_dst1  = (const float*)d_in[7];
  const float* b1      = (const float*)d_in[8];
  const float* ln1_w   = (const float*)d_in[9];
  const float* ln1_b   = (const float*)d_in[10];
  const float* W2      = (const float*)d_in[11];
  const float* a_src2  = (const float*)d_in[12];
  const float* a_dst2  = (const float*)d_in[13];
  const float* b2      = (const float*)d_in[14];
  const float* ln2_w   = (const float*)d_in[15];
  const float* ln2_b   = (const float*)d_in[16];
  const float* pw      = (const float*)d_in[17];
  const float* pb      = (const float*)d_in[18];
  const float* bng     = (const float*)d_in[19];
  const float* bnb     = (const float*)d_in[20];
  const float* cw      = (const float*)d_in[21];
  const float* cb      = (const float*)d_in[22];
  float* out = (float*)d_out;

  constexpr int Nn = 20000, Ee = 320000;
  constexpr int Et = Ee + Nn;           // edges incl. self-loops

  // ---- workspace carve-out (~210 MB) ----
  char* wsp = (char*)d_ws;
  auto take = [&](size_t bytes) {
    char* p = wsp;
    wsp += (bytes + 255) & ~(size_t)255;
    return p;
  };
  unsigned short* Abf = (unsigned short*)take((size_t)Nn * 1024 * 2);   // bf16 GEMM A
  unsigned short* Wbf = (unsigned short*)take((size_t)1024 * 1024 * 2); // bf16 weights
  float*    Hf  = (float*)take((size_t)Nn * 1024 * 4);  // GEMM out (h)
  float*    Of  = (float*)take((size_t)Nn * 1024 * 4);  // aggregated out
  float*    as_ = (float*)take((size_t)Nn * 8 * 4);
  float*    ad_ = (float*)take((size_t)Nn * 8 * 4);
  unsigned* mx  = (unsigned*)take((size_t)Nn * 8 * 4);
  float*    sm  = (float*)take((size_t)Nn * 8 * 4);
  float* pooled = (float*)take(16 * 128 * 4);
  float* cnt    = (float*)take(16 * 4);
  float* hfin   = (float*)Abf;   // reuse after GEMM2 consumed Abf

  // ===== input layer norm -> bf16 =====
  ln_kernel<<<Nn, 256, 0, stream>>>(x, nullptr, ln_in_w, ln_in_b, 1024, 0, nullptr, Abf);

  // ===== GAT layer 1 (8 heads x 128) =====
  f32_to_bf16<<<2048, 256, 0, stream>>>(W1, Wbf, (size_t)1024 * 1024);
  gemm_bf16_wmma<8, 1024><<<Nn / 16, 256, 0, stream>>>(Abf, Wbf, Hf);

  fill_f32<<<4096, 256, 0, stream>>>(Of, 0.f, (size_t)Nn * 1024);
  fill_f32<<<640, 256, 0, stream>>>(sm, 0.f, (size_t)Nn * 8);
  fill_u32<<<640, 256, 0, stream>>>(mx, NEG_INF_KEY, (size_t)Nn * 8);

  attn_coef<<<(Nn * 8 + 7) / 8, 256, 0, stream>>>(Hf, a_src1, a_dst1, Nn, 8, 128, as_, ad_);
  edge_max <<<(Et * 8 + 255) / 256, 256, 0, stream>>>(ei, Ee, Nn, 8, as_, ad_, mx);
  edge_sum <<<(Et * 8 + 255) / 256, 256, 0, stream>>>(ei, Ee, Nn, 8, as_, ad_, mx, sm);
  edge_aggr<<<(Et * 8 + 7) / 8, 256, 0, stream>>>(ei, Ee, Nn, 8, 128, as_, ad_, mx, sm, Hf, Of);

  // +b1, LN, ReLU -> bf16 for GEMM2
  ln_kernel<<<Nn, 256, 0, stream>>>(Of, b1, ln1_w, ln1_b, 1024, 1, nullptr, Abf);

  // ===== GAT layer 2 (1 head x 128) =====
  f32_to_bf16<<<512, 256, 0, stream>>>(W2, Wbf, (size_t)128 * 1024);
  gemm_bf16_wmma<1, 1024><<<Nn / 16, 256, 0, stream>>>(Abf, Wbf, Hf);

  fill_f32<<<512, 256, 0, stream>>>(Of, 0.f, (size_t)Nn * 128);
  fill_f32<<<80, 256, 0, stream>>>(sm, 0.f, (size_t)Nn);
  fill_u32<<<80, 256, 0, stream>>>(mx, NEG_INF_KEY, (size_t)Nn);

  attn_coef<<<(Nn + 7) / 8, 256, 0, stream>>>(Hf, a_src2, a_dst2, Nn, 1, 128, as_, ad_);
  edge_max <<<(Et + 255) / 256, 256, 0, stream>>>(ei, Ee, Nn, 1, as_, ad_, mx);
  edge_sum <<<(Et + 255) / 256, 256, 0, stream>>>(ei, Ee, Nn, 1, as_, ad_, mx, sm);
  edge_aggr<<<(Et + 7) / 8, 256, 0, stream>>>(ei, Ee, Nn, 1, 128, as_, ad_, mx, sm, Hf, Of);

  // +b2, LN, ReLU -> f32 final node features
  ln_kernel<<<Nn, 256, 0, stream>>>(Of, b2, ln2_w, ln2_b, 128, 1, hfin, nullptr);

  // ===== pooling + hyperbolic head =====
  fill_f32<<<8, 256, 0, stream>>>(pooled, 0.f, (size_t)16 * 128);
  fill_f32<<<1, 32, 0, stream>>>(cnt, 0.f, (size_t)16);
  pool_kernel<<<(Nn + 7) / 8, 256, 0, stream>>>(hfin, batch, pooled, cnt, Nn);
  head_kernel<<<1, 512, 0, stream>>>(pooled, cnt, pw, pb, bng, bnb, cw, cb, out);
}
